// SparseBayesianLinear_1073741824313
// MI455X (gfx1250) — matrix operand order repaired and checked by reading, not compiled, not run on hardware
//
#include <hip/hip_runtime.h>

// ---------------------------------------------------------------------------
// SparseBayesianLinear fused dual-GEMM for MI455X (gfx1250, wave32, WMMA).
//
// scores = (x @ keys^T) / 64          keys = mu_w * softplus(sigma_w)
// raw    = relu(scores - gate)
// out    = (x @ mu_w^T) * raw + bias
// Outputs (flat, in order): final [8192*4096], scores [8192*4096],
//                           masked [8192*4096]  (final == masked)
//
// Compute-bound (5.5e11 FLOP vs ~0.67 GB minimum traffic) -> bf16 WMMA path.
// bf16 copies of x / mu / keys total 134 MB -> GEMM working set fits in the
// 192 MB L2 after the conversion pass.
//
// Staging: async global->LDS copies (ASYNCcnt) with double-buffered LDS when
// the gfx1250 async builtins are available; falls back to register staging.
// ---------------------------------------------------------------------------

#define IN_F   4096
#define OUT_F  4096
#define NROWS  8192   // 4 * 2048

#define BM 128
#define BN 64
#define BK 32
#define NKT (IN_F / BK)   // 128 K-steps

typedef __attribute__((ext_vector_type(16))) __bf16 v16bf;
typedef __attribute__((ext_vector_type(8)))  float  v8f;
typedef __attribute__((ext_vector_type(4)))  int    v4i;

#if defined(__gfx1250__) && __has_builtin(__builtin_amdgcn_global_load_async_to_lds_b128)
#define USE_ASYNC_LDS 1
#else
#define USE_ASYNC_LDS 0
#endif

union FragBF {
    struct { uint4 lo, hi; } q;   // two 16B chunks = 16 bf16 elements
    v16bf v;
};

__device__ __forceinline__ unsigned short f32_to_bf16(float f) {
    union { float f; unsigned int u; } c; c.f = f;
    unsigned int u = c.u;
    u += 0x7FFFu + ((u >> 16) & 1u);          // round-to-nearest-even
    return (unsigned short)(u >> 16);
}

#if USE_ASYNC_LDS
typedef __attribute__((address_space(1))) v4i GV4;   // global int4*
typedef __attribute__((address_space(3))) v4i LV4;   // LDS    int4*

__device__ __forceinline__ void async_cp16(const void* g, void* l) {
    // GLOBAL_LOAD_ASYNC_TO_LDS_B128: LDS[l] = MEM[g], tracked by ASYNCcnt
    __builtin_amdgcn_global_load_async_to_lds_b128((GV4*)g, (LV4*)l, 0, 0);
}

__device__ __forceinline__ void wait_async0() {
#if __has_builtin(__builtin_amdgcn_s_wait_asynccnt)
    __builtin_amdgcn_s_wait_asynccnt(0);
#else
    asm volatile("s_wait_asynccnt 0x0" ::: "memory");
#endif
}
#endif

// --------------------------- conversion kernels ----------------------------

__global__ void convert_x_bf16(const float* __restrict__ x,
                               unsigned short* __restrict__ xb, int n4) {
    const int stride = gridDim.x * blockDim.x;
    for (int i = blockIdx.x * blockDim.x + threadIdx.x; i < n4; i += stride) {
        float4 v = ((const float4*)x)[i];
        ushort4 o;
        o.x = f32_to_bf16(v.x); o.y = f32_to_bf16(v.y);
        o.z = f32_to_bf16(v.z); o.w = f32_to_bf16(v.w);
        ((ushort4*)xb)[i] = o;
    }
}

__device__ __forceinline__ float softplus_f(float z) {
    return (z > 0.f) ? (z + log1pf(__expf(-z))) : log1pf(__expf(z));
}

__global__ void convert_w_bf16(const float* __restrict__ mu,
                               const float* __restrict__ sig,
                               unsigned short* __restrict__ mub,
                               unsigned short* __restrict__ keyb, int n4) {
    const int stride = gridDim.x * blockDim.x;
    for (int i = blockIdx.x * blockDim.x + threadIdx.x; i < n4; i += stride) {
        float4 m = ((const float4*)mu)[i];
        float4 s = ((const float4*)sig)[i];
        ushort4 om, ok;
        om.x = f32_to_bf16(m.x); ok.x = f32_to_bf16(m.x * softplus_f(s.x));
        om.y = f32_to_bf16(m.y); ok.y = f32_to_bf16(m.y * softplus_f(s.y));
        om.z = f32_to_bf16(m.z); ok.z = f32_to_bf16(m.z * softplus_f(s.z));
        om.w = f32_to_bf16(m.w); ok.w = f32_to_bf16(m.w * softplus_f(s.w));
        ((ushort4*)mub)[i]  = om;
        ((ushort4*)keyb)[i] = ok;
    }
}

// --------------------------- fused dual GEMM -------------------------------
// Grid: (OUT_F/BN, NROWS/BM) = (64, 64). Block: 256 threads = 8 waves.
// Wave grid 4(M) x 2(N): each wave owns a 32x32 output sub-tile, i.e.
// 2x2 WMMA tiles, each with two f32 accumulators (dense + scores).

__global__ __launch_bounds__(256)
void fused_dual_gemm(const unsigned short* __restrict__ xb,
                     const unsigned short* __restrict__ mub,
                     const unsigned short* __restrict__ keyb,
                     const float* __restrict__ gate,
                     const float* __restrict__ bias,
                     float* __restrict__ out_final,
                     float* __restrict__ out_scores,
                     float* __restrict__ out_masked) {
    const int tid  = threadIdx.x;
    const int lane = tid & 31;
    const int wid  = tid >> 5;
    const int wm   = wid & 3;        // 0..3  -> M offset wm*32
    const int wn   = wid >> 2;       // 0..1  -> N offset wn*32
    const int m0   = blockIdx.y * BM;
    const int n0   = blockIdx.x * BN;

    // cooperative staging: thread -> (row sr, col sc) of a 64x32 chunk
    const int sr = tid >> 2;             // 0..63
    const int sc = (tid & 3) * 8;        // 0,8,16,24

    const int lm = lane & 15;   // fragment row/col within 16
    const int lh = lane >> 4;   // half-wave selector

    v8f accS[2][2] = {};   // scores accumulators
    v8f accD[2][2] = {};   // dense  accumulators

    // per-thread global staging pointers (advance by kt*BK elements)
    const unsigned short* gA0 = xb   + (size_t)(m0 + sr) * IN_F + sc;
    const unsigned short* gA1 = gA0  + (size_t)64 * IN_F;
    const unsigned short* gBm = mub  + (size_t)(n0 + sr) * IN_F + sc;
    const unsigned short* gBk = keyb + (size_t)(n0 + sr) * IN_F + sc;

#if USE_ASYNC_LDS
    // ---- async global->LDS pipeline, double-buffered ----
    __shared__ __align__(16) unsigned short ldsA [2][BM * BK];   // 2 x 8 KB
    __shared__ __align__(16) unsigned short ldsBm[2][BN * BK];   // 2 x 4 KB
    __shared__ __align__(16) unsigned short ldsBk[2][BN * BK];   // 2 x 4 KB

    // kick off tile 0 into buffer 0
    async_cp16(gA0, &ldsA[0][sr * BK + sc]);
    async_cp16(gA1, &ldsA[0][(64 + sr) * BK + sc]);
    async_cp16(gBm, &ldsBm[0][sr * BK + sc]);
    async_cp16(gBk, &ldsBk[0][sr * BK + sc]);

    for (int kt = 0; kt < NKT; ++kt) {
        wait_async0();          // this wave's copies for tile kt are in LDS
        __syncthreads();        // everyone's copies are in LDS; prev reads done

        const int cur = kt & 1;
        if (kt + 1 < NKT) {     // next tile in flight during the WMMAs below
            const int ko = (kt + 1) * BK;
            const int nxt = cur ^ 1;
            async_cp16(gA0 + ko, &ldsA[nxt][sr * BK + sc]);
            async_cp16(gA1 + ko, &ldsA[nxt][(64 + sr) * BK + sc]);
            async_cp16(gBm + ko, &ldsBm[nxt][sr * BK + sc]);
            async_cp16(gBk + ko, &ldsBk[nxt][sr * BK + sc]);
            if (kt + 2 < NKT) {                // warm L2 two tiles ahead
                const int k2 = (kt + 2) * BK;
                __builtin_prefetch(gA0 + k2, 0, 3);
                __builtin_prefetch(gBm + k2, 0, 3);
                __builtin_prefetch(gBk + k2, 0, 3);
            }
        }

        // ---- fragment loads from LDS (ISA 16-bit A / B layouts) ----
        FragBF a[2], bm[2], bk[2];
#pragma unroll
        for (int i = 0; i < 2; ++i) {
            const unsigned short* p = &ldsA[cur][(wm * 32 + i * 16 + lm) * BK];
            a[i].q.lo = *(const uint4*)(p + lh * 8);
            a[i].q.hi = *(const uint4*)(p + 16 + lh * 8);
        }
#pragma unroll
        for (int j = 0; j < 2; ++j) {
            const int br = (wn * 32 + j * 16 + lm) * BK + lh * 8;
            bm[j].q.lo = *(const uint4*)&ldsBm[cur][br];
            bm[j].q.hi = *(const uint4*)&ldsBm[cur][br + 16];
            bk[j].q.lo = *(const uint4*)&ldsBk[cur][br];
            bk[j].q.hi = *(const uint4*)&ldsBk[cur][br + 16];
        }

#pragma unroll
        for (int i = 0; i < 2; ++i)
#pragma unroll
            for (int j = 0; j < 2; ++j) {
                accD[i][j] = __builtin_amdgcn_wmma_f32_16x16x32_bf16(
                    false, a[i].v, false, bm[j].v, (short)0, accD[i][j], false, false);
                accS[i][j] = __builtin_amdgcn_wmma_f32_16x16x32_bf16(
                    false, a[i].v, false, bk[j].v, (short)0, accS[i][j], false, false);
            }
    }
#else
    // ---- fallback: register staging, single-buffered LDS ----
    __shared__ __align__(16) unsigned short ldsA [BM * BK];
    __shared__ __align__(16) unsigned short ldsBm[BN * BK];
    __shared__ __align__(16) unsigned short ldsBk[BN * BK];

    uint4 pa0 = *(const uint4*)gA0;
    uint4 pa1 = *(const uint4*)gA1;
    uint4 pbm = *(const uint4*)gBm;
    uint4 pbk = *(const uint4*)gBk;

    for (int kt = 0; kt < NKT; ++kt) {
        __syncthreads();
        *(uint4*)&ldsA [sr * BK + sc]        = pa0;
        *(uint4*)&ldsA [(64 + sr) * BK + sc] = pa1;
        *(uint4*)&ldsBm[sr * BK + sc]        = pbm;
        *(uint4*)&ldsBk[sr * BK + sc]        = pbk;
        __syncthreads();

        if (kt + 1 < NKT) {
            const int ko = (kt + 1) * BK;
            pa0 = *(const uint4*)(gA0 + ko);
            pa1 = *(const uint4*)(gA1 + ko);
            pbm = *(const uint4*)(gBm + ko);
            pbk = *(const uint4*)(gBk + ko);
            if (kt + 2 < NKT) {
                const int k2 = (kt + 2) * BK;
                __builtin_prefetch(gA0 + k2, 0, 3);
                __builtin_prefetch(gBm + k2, 0, 3);
                __builtin_prefetch(gBk + k2, 0, 3);
            }
        }

        FragBF a[2], bm[2], bk[2];
#pragma unroll
        for (int i = 0; i < 2; ++i) {
            const unsigned short* p = &ldsA[(wm * 32 + i * 16 + lm) * BK];
            a[i].q.lo = *(const uint4*)(p + lh * 8);
            a[i].q.hi = *(const uint4*)(p + 16 + lh * 8);
        }
#pragma unroll
        for (int j = 0; j < 2; ++j) {
            const int br = (wn * 32 + j * 16 + lm) * BK + lh * 8;
            bm[j].q.lo = *(const uint4*)&ldsBm[br];
            bm[j].q.hi = *(const uint4*)&ldsBm[br + 16];
            bk[j].q.lo = *(const uint4*)&ldsBk[br];
            bk[j].q.hi = *(const uint4*)&ldsBk[br + 16];
        }

#pragma unroll
        for (int i = 0; i < 2; ++i)
#pragma unroll
            for (int j = 0; j < 2; ++j) {
                accD[i][j] = __builtin_amdgcn_wmma_f32_16x16x32_bf16(
                    false, a[i].v, false, bm[j].v, (short)0, accD[i][j], false, false);
                accS[i][j] = __builtin_amdgcn_wmma_f32_16x16x32_bf16(
                    false, a[i].v, false, bk[j].v, (short)0, accS[i][j], false, false);
            }
    }
#endif

    // ---- fused epilogue: scale, gate-ReLU, mask, bias, 3 outputs ----
    const float scale = 0.015625f;   // 1/sqrt(4096)
#pragma unroll
    for (int j = 0; j < 2; ++j) {
        const int gn = n0 + wn * 32 + j * 16 + lm;     // C/D layout: N = lane%16
        const float g = gate[gn];
        const float b = bias[gn];
#pragma unroll
        for (int i = 0; i < 2; ++i) {
            const int rowb = m0 + wm * 32 + i * 16 + lh * 8;  // M = r + 8*(lane/16)
#pragma unroll
            for (int r = 0; r < 8; ++r) {
                const size_t off = (size_t)(rowb + r) * OUT_F + gn;
                float s   = accS[i][j][r] * scale;
                float raw = s - g; raw = raw > 0.f ? raw : 0.f;
                float f   = accD[i][j][r] * raw + b;
                out_scores[off] = s;
                out_final[off]  = f;
                out_masked[off] = f;
            }
        }
    }
}

// ------------------------------- launcher ----------------------------------

extern "C" void kernel_launch(void* const* d_in, const int* in_sizes, int n_in,
                              void* d_out, int out_size, void* d_ws, size_t ws_size,
                              hipStream_t stream) {
    (void)in_sizes; (void)n_in; (void)out_size; (void)ws_size;

    const float* x    = (const float*)d_in[0];
    const float* mu   = (const float*)d_in[1];
    const float* sig  = (const float*)d_in[2];
    const float* gate = (const float*)d_in[3];
    const float* bias = (const float*)d_in[4];
    float* out = (float*)d_out;

    // workspace layout (bf16 as u16): x [8192*4096], mu [4096*4096], keys [4096*4096]
    unsigned short* xb  = (unsigned short*)d_ws;
    unsigned short* mub = xb  + (size_t)NROWS * IN_F;
    unsigned short* kyb = mub + (size_t)OUT_F * IN_F;

    convert_x_bf16<<<2048, 256, 0, stream>>>(x, xb, (NROWS * IN_F) / 4);
    convert_w_bf16<<<2048, 256, 0, stream>>>(mu, sig, mub, kyb, (OUT_F * IN_F) / 4);

    dim3 grid(OUT_F / BN, NROWS / BM);   // (64, 64)
    fused_dual_gemm<<<grid, 256, 0, stream>>>(
        xb, mub, kyb, gate, bias,
        out,                                   // final_output
        out + (size_t)NROWS * OUT_F,           // scores
        out + 2 * (size_t)NROWS * OUT_F);      // masked_output
}
